// GAT_model_88046829568493
// MI455X (gfx1250) — compile-verified, matrix operand order checked by reading, not run on hardware
//
#include <hip/hip_runtime.h>
#include <hip/hip_bf16.h>

// ---------------------------------------------------------------------------
// GAT model on MI455X (gfx1250, wave32, WMMA).
// Dense GEMMs: operands pre-packed to bf16 in WMMA fragment-major layout so
// the inner loop is pure b128 loads + v_wmma_f32_16x16x32_bf16 (no predication).
// Edge attention is contracted early (edge_w x att_edge -> 3x64 vectors),
// removing the 850k x 64 x 192 edge GEMM entirely.
// ---------------------------------------------------------------------------

typedef __bf16 bf16x16 __attribute__((ext_vector_type(16)));
typedef float  f32x8   __attribute__((ext_vector_type(8)));

#define N_NODES 50000
#define N_EDGES 800000
#define N_EAUG  850000
#define N_GRAPHS 512

// ---- order-preserving float <-> uint for atomicMax-based segment_max ------
__device__ __forceinline__ unsigned f2ord(float f) {
    int i = __float_as_int(f);
    return (unsigned)(i ^ ((i >> 31) | 0x80000000));
}
__device__ __forceinline__ float ord2f(unsigned u) {
    int i = (u & 0x80000000u) ? (int)(u & 0x7FFFFFFFu) : ~(int)u;
    return __int_as_float(i);
}

// ---------------------------------------------------------------------------
// Pack f32 A (M x K, row stride lda) into bf16 WMMA A-fragment-major layout.
// Fragment tile (tm, tk) = 16 rows x 32 k. Lane l of the consuming wave reads
// 32 contiguous bytes at ((tm*ntk + tk)*32 + l)*16 bf16.
// Element e of lane l -> A[tm*16 + (l&15)][tk*32 + e + (e&8) + 8*(l>>4)].
// Addresses are clamped (not branched) so no exec-divergent loads are emitted.
// ---------------------------------------------------------------------------
__global__ void pack_a_kernel(const float* __restrict__ A, int lda,
                              __bf16* __restrict__ out,
                              int M, int K, int ntm, int ntk)
{
    int idx = blockIdx.x * blockDim.x + threadIdx.x;   // (tile, lane)
    if (idx >= ntm * ntk * 32) return;
    int lane = idx & 31;
    int tile = idx >> 5;
    int tk = tile % ntk;
    int tm = tile / ntk;
    int m = tm * 16 + (lane & 15);
    int half = lane >> 4;
    int mc = m < M ? m : M - 1;
    bool mok = m < M;
    __bf16* dst = out + (size_t)idx * 16;
#pragma unroll
    for (int e = 0; e < 16; ++e) {
        int k = tk * 32 + e + (e & 8) + 8 * half;
        int kc = k < K ? k : K - 1;
        float v = A[(size_t)mc * lda + kc];
        v = (mok && k < K) ? v : 0.f;
        dst[e] = (__bf16)v;
    }
}

// Pack weight W (K x N row-major) into bf16 WMMA B-fragment-major layout
// (fold of transpose + conversion). Lane l = column tn*16 + (l&15).
__global__ void pack_b_kernel(const float* __restrict__ W,
                              __bf16* __restrict__ out,
                              int K, int N, int ntn, int ntk)
{
    int idx = blockIdx.x * blockDim.x + threadIdx.x;
    if (idx >= ntn * ntk * 32) return;
    int lane = idx & 31;
    int tile = idx >> 5;
    int tk = tile % ntk;
    int tn = tile / ntk;
    int n = tn * 16 + (lane & 15);
    int half = lane >> 4;
    __bf16* dst = out + (size_t)idx * 16;
#pragma unroll
    for (int e = 0; e < 16; ++e) {
        int k = tk * 32 + e + (e & 8) + 8 * half;
        int kc = k < K ? k : K - 1;
        float v = W[(size_t)kc * N + n];
        v = (k < K) ? v : 0.f;
        dst[e] = (__bf16)v;
    }
}

// ---------------------------------------------------------------------------
// Packed bf16 WMMA GEMM: C[M,N] = act(A*B + bias).
// Block = 128 threads = 4 waves; each wave owns one 16-col tile and TWO
// 16-row tiles (B fragment reused across both wmmas). Inner loop per k-step:
// 3 x 32B coalesced loads + 2 x v_wmma_f32_16x16x32_bf16, no predication.
// ---------------------------------------------------------------------------
__global__ void wmma_gemm_packed(const __bf16* __restrict__ Ap,
                                 const __bf16* __restrict__ Bp,
                                 const float* __restrict__ bias,
                                 float* __restrict__ C, int ldc,
                                 int M, int N, int ntk, int relu)
{
    const int wave = threadIdx.x >> 5;
    const int lane = threadIdx.x & 31;
    const int half = lane >> 4;
    const int l16  = lane & 15;
    const int tileN = blockIdx.x * 4 + wave;           // 16-col tile index
    if (tileN * 16 >= N) return;                       // wave-uniform exit
    const int tm0 = blockIdx.y * 2;                    // two row tiles

    const bf16x16* A0 = (const bf16x16*)Ap + (size_t)tm0 * ntk * 32 + lane;
    const bf16x16* A1 = A0 + (size_t)ntk * 32;
    const bf16x16* Bw = (const bf16x16*)Bp + (size_t)tileN * ntk * 32 + lane;

    f32x8 acc0 = {0.f, 0.f, 0.f, 0.f, 0.f, 0.f, 0.f, 0.f};
    f32x8 acc1 = {0.f, 0.f, 0.f, 0.f, 0.f, 0.f, 0.f, 0.f};

    for (int tk = 0; tk < ntk; ++tk) {
        __builtin_prefetch(&A0[(tk + 1) * 32], 0, 1);  // global_prefetch_b8
        __builtin_prefetch(&A1[(tk + 1) * 32], 0, 1);
        bf16x16 a0 = A0[tk * 32];
        bf16x16 a1 = A1[tk * 32];
        bf16x16 b  = Bw[tk * 32];
        acc0 = __builtin_amdgcn_wmma_f32_16x16x32_bf16(
                   false, a0, false, b, (short)0, acc0, false, false);
        acc1 = __builtin_amdgcn_wmma_f32_16x16x32_bf16(
                   false, a1, false, b, (short)0, acc1, false, false);
    }

    const int col = tileN * 16 + l16;
#pragma unroll
    for (int r = 0; r < 8; ++r) {
        int row0 = tm0 * 16 + r + 8 * half;
        int row1 = row0 + 16;
        float v0 = acc0[r], v1 = acc1[r];
        if (bias) { float bv = bias[col]; v0 += bv; v1 += bv; }
        if (relu) { v0 = fmaxf(v0, 0.f); v1 = fmaxf(v1, 0.f); }
        if (row0 < M) C[(size_t)row0 * ldc + col] = v0;
        if (row1 < M) C[(size_t)row1 * ldc + col] = v1;
    }
}

// ---- small helpers --------------------------------------------------------
__global__ void fillf_kernel(float* p, float v, int n) {
    int i = blockIdx.x * blockDim.x + threadIdx.x;
    if (i < n) p[i] = v;
}
__global__ void fillu_kernel(unsigned* p, unsigned v, int n) {
    int i = blockIdx.x * blockDim.x + threadIdx.x;
    if (i < n) p[i] = v;
}

// column mean of ea (E x 64) -> ea_mean[64]; one block per column
__global__ void col_mean_kernel(const float* __restrict__ ea, float* __restrict__ mean,
                                int rows, int cols) {
    int c = blockIdx.x;
    float s = 0.f;
    for (int r = threadIdx.x; r < rows; r += blockDim.x)
        s += ea[(size_t)r * cols + c];
    __shared__ float red[256];
    red[threadIdx.x] = s;
    __syncthreads();
    for (int o = 128; o > 0; o >>= 1) {
        if (threadIdx.x < o) red[threadIdx.x] += red[threadIdx.x + o];
        __syncthreads();
    }
    if (threadIdx.x == 0) mean[c] = red[0] / (float)rows;
}

// v_edge[h*64+k] = sum_c edge_w[k, h*64+c] * att_edge[h,c]   (early contraction)
__global__ void edge_att_contract_kernel(const float* __restrict__ edge_w,
                                         const float* __restrict__ att_edge,
                                         float* __restrict__ v_edge) {
    int t = threadIdx.x;
    if (t >= 192) return;
    int h = t / 64, k = t % 64;
    float s = 0.f;
    for (int c = 0; c < 64; ++c)
        s += edge_w[k * 192 + h * 64 + c] * att_edge[h * 64 + c];
    v_edge[h * 64 + k] = s;
}

// per-node attention logits a_src/a_dst from transformed features ht (N x 192)
__global__ void node_att_kernel(const float* __restrict__ ht,
                                const float* __restrict__ att_src,
                                const float* __restrict__ att_dst,
                                float* __restrict__ a_src, float* __restrict__ a_dst,
                                int n_nodes) {
    int n = blockIdx.x * blockDim.x + threadIdx.x;
    if (n >= n_nodes) return;
    const float4* hp = (const float4*)(ht + (size_t)n * 192);
    const float4* as = (const float4*)att_src;
    const float4* ad = (const float4*)att_dst;
#pragma unroll
    for (int h = 0; h < 3; ++h) {
        float s = 0.f, d = 0.f;
#pragma unroll
        for (int q = 0; q < 16; ++q) {
            float4 v = hp[h * 16 + q];
            float4 a = as[h * 16 + q];
            float4 b = ad[h * 16 + q];
            s += v.x * a.x + v.y * a.y + v.z * a.z + v.w * a.w;
            d += v.x * b.x + v.y * b.y + v.z * b.z + v.w * b.w;
        }
        a_src[n * 3 + h] = s;
        a_dst[n * 3 + h] = d;
    }
}

// stage 1: raw logits + leaky-relu + segment max (dst) via ordered atomicMax
__global__ void edge_logits_kernel(const int* __restrict__ ei,
                                   const float* __restrict__ ea,
                                   const float* __restrict__ ea_mean,
                                   const float* __restrict__ v_edge,
                                   const float* __restrict__ a_src,
                                   const float* __restrict__ a_dst,
                                   float* __restrict__ alpha,
                                   unsigned* __restrict__ amax_u) {
    int e = blockIdx.x * blockDim.x + threadIdx.x;
    if (e >= N_EAUG) return;
    int src, dst;
    const float* row;
    if (e < N_EDGES) { src = ei[e]; dst = ei[N_EDGES + e]; row = ea + (size_t)e * 64; }
    else             { src = dst = e - N_EDGES;            row = ea_mean; }
    const float4* r4 = (const float4*)row;
    const float4* v4 = (const float4*)v_edge;
    float ae0 = 0.f, ae1 = 0.f, ae2 = 0.f;
#pragma unroll
    for (int q = 0; q < 16; ++q) {
        float4 xv = r4[q];
        float4 w0 = v4[q], w1 = v4[16 + q], w2 = v4[32 + q];
        ae0 += xv.x * w0.x + xv.y * w0.y + xv.z * w0.z + xv.w * w0.w;
        ae1 += xv.x * w1.x + xv.y * w1.y + xv.z * w1.z + xv.w * w1.w;
        ae2 += xv.x * w2.x + xv.y * w2.y + xv.z * w2.z + xv.w * w2.w;
    }
    float ae[3] = {ae0, ae1, ae2};
#pragma unroll
    for (int h = 0; h < 3; ++h) {
        float l = a_src[src * 3 + h] + a_dst[dst * 3 + h] + ae[h];
        l = (l > 0.f) ? l : 0.2f * l;             // leaky_relu 0.2
        alpha[(size_t)e * 3 + h] = l;
        atomicMax(&amax_u[dst * 3 + h], f2ord(l));
    }
}

// stage 2: exp(alpha - amax[dst]) and segment sum of denominators
__global__ void edge_exp_kernel(const int* __restrict__ ei,
                                float* __restrict__ alpha,
                                const unsigned* __restrict__ amax_u,
                                float* __restrict__ denom) {
    int e = blockIdx.x * blockDim.x + threadIdx.x;
    if (e >= N_EAUG) return;
    int dst = (e < N_EDGES) ? ei[N_EDGES + e] : e - N_EDGES;
#pragma unroll
    for (int h = 0; h < 3; ++h) {
        float m = ord2f(amax_u[dst * 3 + h]);
        float p = __expf(alpha[(size_t)e * 3 + h] - m);
        alpha[(size_t)e * 3 + h] = p;
        atomicAdd(&denom[dst * 3 + h], p);
    }
}

// stage 3: wave-per-edge weighted scatter of h[src] into accum[dst] (192 ch)
__global__ void edge_scatter_kernel(const int* __restrict__ ei,
                                    const float* __restrict__ alpha,
                                    const float* __restrict__ denom,
                                    const float* __restrict__ ht,
                                    float* __restrict__ accum) {
    int e = blockIdx.x * (blockDim.x >> 5) + (threadIdx.x >> 5);
    if (e >= N_EAUG) return;
    int lane = threadIdx.x & 31;
    int src, dst;
    if (e < N_EDGES) { src = ei[e]; dst = ei[N_EDGES + e]; }
    else             { src = dst = e - N_EDGES; }
    float w[3];
#pragma unroll
    for (int h = 0; h < 3; ++h)
        w[h] = alpha[(size_t)e * 3 + h] / (denom[dst * 3 + h] + 1e-16f);
#pragma unroll
    for (int t = lane; t < 192; t += 32) {
        int h = t >> 6;
        atomicAdd(&accum[(size_t)dst * 192 + t], w[h] * ht[(size_t)src * 192 + t]);
    }
}

// head-mean + bias + relu -> next layer features (N x 64)
__global__ void finalize_kernel(const float* __restrict__ accum,
                                const float* __restrict__ bias,
                                float* __restrict__ hout) {
    int idx = blockIdx.x * blockDim.x + threadIdx.x;
    if (idx >= N_NODES * 64) return;
    int n = idx >> 6, c = idx & 63;
    const float* a = accum + (size_t)n * 192;
    float v = (a[c] + a[64 + c] + a[128 + c]) * (1.f / 3.f) + bias[c];
    hout[idx] = v > 0.f ? v : 0.f;
}

// multipli-scale + batch segment-sum pooling
__global__ void pool_kernel(const float* __restrict__ h, const float* __restrict__ x,
                            const int* __restrict__ batch, float* __restrict__ pooled) {
    int idx = blockIdx.x * blockDim.x + threadIdx.x;
    if (idx >= N_NODES * 64) return;
    int n = idx >> 6, c = idx & 63;
    atomicAdd(&pooled[batch[n] * 64 + c], h[idx] * x[(size_t)n * 93 + 92]);
}

// final 128 -> 1 projection + relu
__global__ void mlp2_kernel(const float* __restrict__ g1, const float* __restrict__ w,
                            const float* __restrict__ b, float* __restrict__ out) {
    int g = blockIdx.x * blockDim.x + threadIdx.x;
    if (g >= N_GRAPHS) return;
    float s = b[0];
#pragma unroll 8
    for (int k = 0; k < 128; ++k) s += g1[g * 128 + k] * w[k];
    out[g] = s > 0.f ? s : 0.f;
}

// ---------------------------------------------------------------------------
extern "C" void kernel_launch(void* const* d_in, const int* in_sizes, int n_in,
                              void* d_out, int out_size, void* d_ws, size_t ws_size,
                              hipStream_t stream) {
    (void)in_sizes; (void)n_in; (void)out_size; (void)ws_size;

    const float* x   = (const float*)d_in[0];     // 50000 x 93
    const float* eat = (const float*)d_in[1];     // 800000 x 50
    const int*   ei  = (const int*)d_in[2];       // 2 x 800000
    const int*   bat = (const int*)d_in[3];       // 50000
    const float* exp_w  = (const float*)d_in[4];
    const float* exp_b  = (const float*)d_in[5];
    const float* eexp_w = (const float*)d_in[6];
    const float* eexp_b = (const float*)d_in[7];
    // layers at 8 + i*6 : lin_w, edge_w, att_src, att_dst, att_edge, bias
    const float* mlp1_w = (const float*)d_in[26];
    const float* mlp1_b = (const float*)d_in[27];
    const float* mlp2_w = (const float*)d_in[28];
    const float* mlp2_b = (const float*)d_in[29];
    float* out = (float*)d_out;

    auto cdiv = [](int a, int b) { return (a + b - 1) / b; };

    // tile counts (two 16-row tiles per wave -> M padded to 32)
    const int ntmN = 2 * cdiv(N_NODES, 32);   // 3126 row tiles for node mats
    const int ntmE = 2 * cdiv(N_EDGES, 32);   // 50000 row tiles for edge mat
    const int ntmG = 2 * cdiv(N_GRAPHS, 32);  // 32

    // ---- workspace carve-up (64B-aligned float units) ----
    float* ws = (float*)d_ws;
    size_t off = 0;
    auto alloc = [&](size_t n) { float* p = ws + off; off += (n + 15) & ~(size_t)15; return p; };
    float* ea      = alloc((size_t)N_EDGES * 64);   // relu(edge_attr @ eexp_w)
    float* h_a     = alloc((size_t)N_NODES * 128);  // ping
    float* h_b     = alloc((size_t)N_NODES * 64);   // pong
    float* ht      = alloc((size_t)N_NODES * 192);  // h @ lin_w
    float* accum   = alloc((size_t)N_NODES * 192);  // scatter accumulator
    float* a_src   = alloc((size_t)N_NODES * 3);
    float* a_dst   = alloc((size_t)N_NODES * 3);
    unsigned* amax = (unsigned*)alloc((size_t)N_NODES * 3);
    float* denom   = alloc((size_t)N_NODES * 3);
    float* alpha   = alloc((size_t)N_EAUG * 3);
    float* ea_mean = alloc(64);
    float* v_edge  = alloc(192);
    float* pooled  = alloc((size_t)N_GRAPHS * 64);
    float* g1      = alloc((size_t)N_GRAPHS * 128);
    // packed bf16 operands (sized in bf16 elems -> /2 for float units)
    __bf16* xp   = (__bf16*)alloc((size_t)ntmN * 3 * 512 / 2);  // x pack (K=92->96)
    __bf16* eap  = (__bf16*)alloc((size_t)ntmE * 2 * 512 / 2);  // edge_attr pack (K=50->64)
    __bf16* hp   = (__bf16*)alloc((size_t)ntmN * 4 * 512 / 2);  // h pack (K<=128)
    __bf16* gp   = (__bf16*)alloc((size_t)ntmG * 2 * 512 / 2);  // pooled pack (K=64)
    __bf16* bexp = (__bf16*)alloc((size_t)8 * 3 * 512 / 2);     // exp_w pack
    __bf16* bee  = (__bf16*)alloc((size_t)4 * 2 * 512 / 2);     // eexp_w pack
    __bf16* blin = (__bf16*)alloc((size_t)12 * 4 * 512 / 2);    // lin_w pack (max)
    __bf16* bm1  = (__bf16*)alloc((size_t)8 * 2 * 512 / 2);     // mlp1_w pack

    const dim3 B256(256), B128(128);

    // ---- weight packs (transpose folded into B-fragment pack) ----
    pack_b_kernel<<<cdiv(8 * 3 * 32, 256), B256, 0, stream>>>(exp_w, bexp, 92, 128, 8, 3);
    pack_b_kernel<<<cdiv(4 * 2 * 32, 256), B256, 0, stream>>>(eexp_w, bee, 50, 64, 4, 2);
    pack_b_kernel<<<cdiv(8 * 2 * 32, 256), B256, 0, stream>>>(mlp1_w, bm1, 64, 128, 8, 2);

    // ---- feature expansions (WMMA) ----
    pack_a_kernel<<<cdiv(ntmN * 3 * 32, 256), B256, 0, stream>>>(x, 93, xp, N_NODES, 92, ntmN, 3);
    wmma_gemm_packed<<<dim3(cdiv(128, 64), cdiv(N_NODES, 32)), B128, 0, stream>>>(
        xp, bexp, exp_b, h_a, 128, N_NODES, 128, 3, 1);

    pack_a_kernel<<<cdiv(ntmE * 2 * 32, 256), B256, 0, stream>>>(eat, 50, eap, N_EDGES, 50, ntmE, 2);
    wmma_gemm_packed<<<dim3(cdiv(64, 64), cdiv(N_EDGES, 32)), B128, 0, stream>>>(
        eap, bee, eexp_b, ea, 64, N_EDGES, 64, 2, 1);
    col_mean_kernel<<<64, B256, 0, stream>>>(ea, ea_mean, N_EDGES, 64);

    fillf_kernel<<<cdiv(N_GRAPHS * 64, 256), B256, 0, stream>>>(pooled, 0.f, N_GRAPHS * 64);

    // ---- 3 GAT layers ----
    float* h_in = h_a;
    float* h_out = h_b;
    for (int i = 0; i < 3; ++i) {
        const float* lin_w    = (const float*)d_in[8 + i * 6 + 0];
        const float* edge_w   = (const float*)d_in[8 + i * 6 + 1];
        const float* att_src  = (const float*)d_in[8 + i * 6 + 2];
        const float* att_dst  = (const float*)d_in[8 + i * 6 + 3];
        const float* att_edge = (const float*)d_in[8 + i * 6 + 4];
        const float* bias     = (const float*)d_in[8 + i * 6 + 5];
        const int din = (i == 0) ? 128 : 64;
        const int ntk = din / 32;

        pack_b_kernel<<<cdiv(12 * ntk * 32, 256), B256, 0, stream>>>(lin_w, blin, din, 192, 12, ntk);
        pack_a_kernel<<<cdiv(ntmN * ntk * 32, 256), B256, 0, stream>>>(
            h_in, din, hp, N_NODES, din, ntmN, ntk);
        // ht = h_in @ lin_w  (no bias / act)
        wmma_gemm_packed<<<dim3(cdiv(192, 64), cdiv(N_NODES, 32)), B128, 0, stream>>>(
            hp, blin, nullptr, ht, 192, N_NODES, 192, ntk, 0);

        edge_att_contract_kernel<<<1, 192, 0, stream>>>(edge_w, att_edge, v_edge);
        node_att_kernel<<<cdiv(N_NODES, 256), B256, 0, stream>>>(
            ht, att_src, att_dst, a_src, a_dst, N_NODES);

        fillu_kernel<<<cdiv(N_NODES * 3, 256), B256, 0, stream>>>(amax, 0u, N_NODES * 3);
        fillf_kernel<<<cdiv(N_NODES * 3, 256), B256, 0, stream>>>(denom, 0.f, N_NODES * 3);
        fillf_kernel<<<cdiv(N_NODES * 192, 256), B256, 0, stream>>>(accum, 0.f, N_NODES * 192);

        edge_logits_kernel<<<cdiv(N_EAUG, 256), B256, 0, stream>>>(
            ei, ea, ea_mean, v_edge, a_src, a_dst, alpha, amax);
        edge_exp_kernel<<<cdiv(N_EAUG, 256), B256, 0, stream>>>(ei, alpha, amax, denom);
        edge_scatter_kernel<<<cdiv(N_EAUG, 8), B256, 0, stream>>>(
            ei, alpha, denom, ht, accum);
        finalize_kernel<<<cdiv(N_NODES * 64, 256), B256, 0, stream>>>(accum, bias, h_out);

        float* t = h_in; h_in = h_out; h_out = (i == 0) ? h_a : t;  // ping-pong
    }
    float* h_final = h_in;  // output of layer 2

    // ---- pooling + MLP head ----
    pool_kernel<<<cdiv(N_NODES * 64, 256), B256, 0, stream>>>(h_final, x, bat, pooled);
    pack_a_kernel<<<cdiv(ntmG * 2 * 32, 256), B256, 0, stream>>>(
        pooled, 64, gp, N_GRAPHS, 64, ntmG, 2);
    wmma_gemm_packed<<<dim3(cdiv(128, 64), cdiv(N_GRAPHS, 32)), B128, 0, stream>>>(
        gp, bm1, mlp1_b, g1, 128, N_GRAPHS, 128, 2, 1);
    mlp2_kernel<<<cdiv(N_GRAPHS, 256), B256, 0, stream>>>(g1, mlp2_w, mlp2_b, out);
}
